// GCN_61280593379652
// MI455X (gfx1250) — matrix operand order repaired and checked by reading, not compiled, not run on hardware
//
#include <hip/hip_runtime.h>
#include <math.h>

// CDNA5 (gfx1250) wave32 WMMA types
typedef __attribute__((ext_vector_type(16))) __bf16 v16bf;
typedef __attribute__((ext_vector_type(8)))  float  v8f;

// ---------------------------------------------------------------------------
// Elementwise helpers
// ---------------------------------------------------------------------------
__global__ void fill_value(float* __restrict__ p, float v, size_t n) {
    size_t i = (size_t)blockIdx.x * blockDim.x + threadIdx.x;
    if (i < n) p[i] = v;
}

__global__ void count_deg(const int* __restrict__ dst, float* __restrict__ deg, int E) {
    int i = blockIdx.x * blockDim.x + threadIdx.x;
    if (i < E) atomicAdd(&deg[dst[i]], 1.0f);
}

__global__ void rsqrt_inplace(float* __restrict__ p, int n) {
    int i = blockIdx.x * blockDim.x + threadIdx.x;
    if (i < n) p[i] = rsqrtf(p[i]);
}

// h[i] = relu(agg[i] + h[i]*dinv[row]^2 + bias[col]), in place on h
__global__ void finalize_relu(const float* __restrict__ agg, float* __restrict__ h,
                              const float* __restrict__ dinv, const float* __restrict__ bias,
                              int N, int F) {
    size_t i = (size_t)blockIdx.x * blockDim.x + threadIdx.x;
    if (i >= (size_t)N * F) return;
    int row = (int)(i / F);
    int col = (int)(i % F);
    float di = dinv[row];
    float v = agg[i] + h[i] * di * di + bias[col];
    h[i] = v > 0.0f ? v : 0.0f;
}

// ---------------------------------------------------------------------------
// Edge aggregation: one wave32 per edge, 32 lanes x float2 covers F=64.
// Coalesced 256B gather of h[src]; 64 coalesced global_atomic_add_f32 into
// agg[dst] (both 25.6MB buffers are L2-resident on the 192MB L2).
// ---------------------------------------------------------------------------
__global__ void scatter_edges(const float* __restrict__ h, const int* __restrict__ src,
                              const int* __restrict__ dst, const float* __restrict__ dinv,
                              float* __restrict__ agg, int E) {
    int wave = (int)(((size_t)blockIdx.x * blockDim.x + threadIdx.x) >> 5);
    if (wave >= E) return;                 // uniform per wave
    int lane = threadIdx.x & 31;
    int s = src[wave];
    int d = dst[wave];
    float coef = dinv[s] * dinv[d];
    const float2* hs = (const float2*)(h + (size_t)s * 64);
    float2 v = hs[lane];
    float* ad = agg + (size_t)d * 64 + 2 * lane;
    atomicAdd(ad,     v.x * coef);
    atomicAdd(ad + 1, v.y * coef);
}

// ---------------------------------------------------------------------------
// WMMA GEMM: C[N x M] = A[N x K] @ B[K x M] (+ bias), f32 in/out.
// - Whole B staged once per block into LDS as bf16, column-major (Blds[m*K+k])
//   so a B fragment (16 contiguous K per lane, 32B aligned) is ds_load_b128 x2.
// - One wave per (16-row tile, TX-column-tile group): TX=2 fixed at compile
//   time -> exactly 2 accumulators (16 VGPRs), no runtime guards, no spills.
// - K-loop pinned with unroll(1) to stop pipelining blowing up live ranges.
// Fragment layouts used here (A/B k order chosen by this code, same for both; hw = lane>>4):
//   A: element e <- K = k0 + 16*hw + e, row M = lane%16  (contiguous)
//   B: element e <- K = k0 + 16*hw + e, col N = lane%16  (contiguous)
//   C/D: element r -> row M = 8*hw + r, col N = lane%16
// ---------------------------------------------------------------------------
template <int TX>
__global__ void gemm_bf16_wmma(const float* __restrict__ A, const float* __restrict__ B,
                               const float* __restrict__ bias, float* __restrict__ C,
                               int K, int M, int num_waves, int col_groups) {
    extern __shared__ __bf16 Blds[];       // [M][K] column-major: Blds[m*K + k]

    // Cooperative transpose+convert of B into LDS (contiguous LDS writes).
    int total = M * K;
    for (int i = threadIdx.x; i < total; i += blockDim.x) {
        int m = i / K;
        int k = i - m * K;
        Blds[i] = (__bf16)B[(size_t)k * M + m];
    }
    __syncthreads();

    int wave = (int)(((size_t)blockIdx.x * blockDim.x + threadIdx.x) >> 5);
    if (wave >= num_waves) return;         // uniform per wave -> EXEC all-ones below
    int lane = threadIdx.x & 31;
    int hw   = lane >> 4;
    int m16  = lane & 15;
    int rowt = wave / col_groups;
    int cg   = wave - rowt * col_groups;
    int arow = rowt * 16 + m16;            // A row this lane feeds
    int col0 = cg * (TX * 16) + m16;       // first B/C column this lane feeds

    v8f acc[TX];
    #pragma unroll
    for (int t = 0; t < TX; ++t) acc[t] = {};

    const float* ap = A + (size_t)arow * K;
    __builtin_prefetch(ap, 0, 0);          // branch-free global_prefetch_b8

    #pragma unroll 1
    for (int k0 = 0; k0 < K; k0 += 32) {
        // A fragment: two contiguous 8-float runs -> 4x global_load_b128 + cvt_pk
        const float4* r0 = (const float4*)(ap + k0 + 16 * hw);
        const float4* r1 = (const float4*)(ap + k0 + 16 * hw + 8);
        float4 x0 = r0[0], x1 = r0[1], x2 = r1[0], x3 = r1[1];
        v16bf a;
        a[0]  = (__bf16)x0.x; a[1]  = (__bf16)x0.y; a[2]  = (__bf16)x0.z; a[3]  = (__bf16)x0.w;
        a[4]  = (__bf16)x1.x; a[5]  = (__bf16)x1.y; a[6]  = (__bf16)x1.z; a[7]  = (__bf16)x1.w;
        a[8]  = (__bf16)x2.x; a[9]  = (__bf16)x2.y; a[10] = (__bf16)x2.z; a[11] = (__bf16)x2.w;
        a[12] = (__bf16)x3.x; a[13] = (__bf16)x3.y; a[14] = (__bf16)x3.z; a[15] = (__bf16)x3.w;

        #pragma unroll
        for (int t = 0; t < TX; ++t) {
            const v16bf b = *(const v16bf*)(Blds + (size_t)(col0 + 16 * t) * K
                                                 + k0 + 16 * hw);
            acc[t] = __builtin_amdgcn_wmma_f32_16x16x32_bf16(false, a, false, b,
                                                             (short)0, acc[t], false, false);
        }
    }

    #pragma unroll
    for (int t = 0; t < TX; ++t) {
        int bcol = col0 + 16 * t;
        float bv = bias ? bias[bcol] : 0.0f;
        #pragma unroll
        for (int r = 0; r < 8; ++r) {
            int orow = rowt * 16 + 8 * hw + r;
            C[(size_t)orow * M + bcol] = acc[t][r] + bv;
        }
    }
}

// ---------------------------------------------------------------------------
// Softmax over C=32 classes: one wave per row, lane == class. wave32-native.
// ---------------------------------------------------------------------------
__global__ void softmax_rows32(float* __restrict__ out, int N) {
    int wave = (int)(((size_t)blockIdx.x * blockDim.x + threadIdx.x) >> 5);
    if (wave >= N) return;
    int lane = threadIdx.x & 31;
    size_t idx = (size_t)wave * 32 + lane;
    float v = out[idx];
    float m = v;
    #pragma unroll
    for (int off = 16; off > 0; off >>= 1) m = fmaxf(m, __shfl_xor(m, off, 32));
    float e = __expf(v - m);
    float s = e;
    #pragma unroll
    for (int off = 16; off > 0; off >>= 1) s += __shfl_xor(s, off, 32);
    out[idx] = e / s;
}

// ---------------------------------------------------------------------------
// Driver
// ---------------------------------------------------------------------------
extern "C" void kernel_launch(void* const* d_in, const int* in_sizes, int n_in,
                              void* d_out, int out_size, void* d_ws, size_t ws_size,
                              hipStream_t stream) {
    const float* x    = (const float*)d_in[0];
    const int*   ei   = (const int*)d_in[1];
    const float* W1   = (const float*)d_in[2];
    const float* b1   = (const float*)d_in[3];
    const float* W2   = (const float*)d_in[4];
    const float* b2   = (const float*)d_in[5];
    const float* Wout = (const float*)d_in[6];
    const float* bout = (const float*)d_in[7];
    float* out = (float*)d_out;

    const int H    = in_sizes[3];            // 64
    const int C    = in_sizes[7];            // 32
    const int F_IN = in_sizes[2] / H;        // 128
    const int N    = in_sizes[0] / F_IN;     // 100000 (multiple of 16)
    const int E    = in_sizes[1] / 2;        // 1600000
    const int* src = ei;
    const int* dst = ei + E;

    // Workspace: dinv[N] | buf0[N*H] | buf1[N*H]  (~51.6 MB)
    float* dinv = (float*)d_ws;
    float* buf0 = dinv + N;
    float* buf1 = buf0 + (size_t)N * H;

    const int T = 256;
    const size_t NH = (size_t)N * H;
    dim3 blkNH((unsigned)((NH + T - 1) / T));
    dim3 blkN((unsigned)((N + T - 1) / T));
    dim3 blkE((unsigned)((E + T - 1) / T));
    dim3 blkEdgeWaves((unsigned)(((size_t)E * 32 + T - 1) / T));
    dim3 blkRowWaves((unsigned)(((size_t)N * 32 + T - 1) / T));

    const int row_tiles = N / 16;
    const int cgH = (H / 16) / 2;            // column groups of 2 tiles (M=64 -> 2)
    const int cgC = (C / 16) / 2;            // (M=32 -> 1)
    const int wavesH = row_tiles * cgH;
    const int wavesC = row_tiles * cgC;
    dim3 blkGemmH((unsigned)(((size_t)wavesH * 32 + T - 1) / T));
    dim3 blkGemmC((unsigned)(((size_t)wavesC * 32 + T - 1) / T));

    // Degrees: deg = 1 (self loop) + sum over dst; then dinv = rsqrt(deg)
    fill_value<<<blkN, T, 0, stream>>>(dinv, 1.0f, (size_t)N);
    count_deg<<<blkE, T, 0, stream>>>(dst, dinv, E);
    rsqrt_inplace<<<blkN, T, 0, stream>>>(dinv, N);

    // ---- Layer 1: h1 = X @ W1 ; aggregate ; relu(+b1) in place in buf0 ----
    gemm_bf16_wmma<2><<<blkGemmH, T, (size_t)H * F_IN * sizeof(__bf16), stream>>>(
        x, W1, nullptr, buf0, F_IN, H, wavesH, cgH);
    fill_value<<<blkNH, T, 0, stream>>>(buf1, 0.0f, NH);
    scatter_edges<<<blkEdgeWaves, T, 0, stream>>>(buf0, src, dst, dinv, buf1, E);
    finalize_relu<<<blkNH, T, 0, stream>>>(buf1, buf0, dinv, b1, N, H);

    // ---- Layer 2: h2 = h1 @ W2 ; aggregate ; relu(+b2) in place in buf1 ----
    gemm_bf16_wmma<2><<<blkGemmH, T, (size_t)H * H * sizeof(__bf16), stream>>>(
        buf0, W2, nullptr, buf1, H, H, wavesH, cgH);
    fill_value<<<blkNH, T, 0, stream>>>(buf0, 0.0f, NH);
    scatter_edges<<<blkEdgeWaves, T, 0, stream>>>(buf1, src, dst, dinv, buf0, E);
    finalize_relu<<<blkNH, T, 0, stream>>>(buf0, buf1, dinv, b2, N, H);

    // ---- Output layer: logits = h2 @ Wout + bout ; softmax in place ----
    gemm_bf16_wmma<2><<<blkGemmC, T, (size_t)C * H * sizeof(__bf16), stream>>>(
        buf1, Wout, bout, out, H, C, wavesC, cgC);
    softmax_rows32<<<blkRowWaves, T, 0, stream>>>(out, N);
}
